// GNNPredictor_9680856285783
// MI455X (gfx1250) — matrix operand order repaired and checked by reading, not compiled, not run on hardware
//
#include <hip/hip_runtime.h>
#include <hip/hip_bf16.h>

#define HD 128
#define GNUM 64
#define DDOCD 512
#define TOUTD 32
#define NEG_SLOPE 0.2f

typedef __attribute__((ext_vector_type(16))) __bf16 v16bf;
typedef __attribute__((ext_vector_type(8)))  float  v8f;

__device__ __forceinline__ unsigned fenc(float f) {
  unsigned u = __float_as_uint(f);
  return (u & 0x80000000u) ? ~u : (u | 0x80000000u);
}
__device__ __forceinline__ float fdec(unsigned k) {
  return (k & 0x80000000u) ? __uint_as_float(k & 0x7fffffffu)
                           : __uint_as_float(~k);
}

// ---------------------------------------------------------------------------
// Pack a 128x128 f32 weight (row-major W[k][n]) into bf16 WMMA B-fragment
// order: [ks(4)][nt(8)][lane(32)][e(16)];  k = ks*32 + 8*(lane>>4) + e (e<8), +16+(e-8) (e>=8),
// n = nt*16 + (lane&15).  (CDNA5 ISA 7.12.2, 16-bit B 32x16 layout.)
// ---------------------------------------------------------------------------
__global__ void pack_w_kernel(const float* __restrict__ W,
                              __bf16* __restrict__ out) {
  int t = blockIdx.x * blockDim.x + threadIdx.x;
  if (t >= 4 * 8 * 32 * 16) return;
  int e    = t & 15;
  int lane = (t >> 4) & 31;
  int nt   = (t >> 9) & 7;
  int ks   = t >> 12;
  int k = ks * 32 + ((lane >> 4) << 3) + (e < 8 ? e : e + 8);
  int n = nt * 16 + (lane & 15);
  out[t] = (__bf16)W[k * HD + n];
}

// ---------------------------------------------------------------------------
// H = X(nrows x 128) @ W(128x128).  One wave computes a 16x128 output tile
// with 8 f32 accumulators; 4 K-steps of v_wmma_f32_16x16x32_bf16.
// A-fragment (16-bit A 16x32 layout): lane L holds row M=L&15,
// elems 0..7 -> K = 8*(L>>4)+0..7, elems 8..15 -> K = 16 + 8*(L>>4)+0..7.
// ---------------------------------------------------------------------------
__global__ void __launch_bounds__(256) gemm128_wmma_kernel(
    const float* __restrict__ X, const __bf16* __restrict__ Bp,
    float* __restrict__ Hout, int nrows) {
  int lane  = threadIdx.x & 31;
  int wave  = threadIdx.x >> 5;
  int tile  = blockIdx.x * 8 + wave;
  int ntile = (nrows + 15) >> 4;
  if (tile >= ntile) return;              // wave-uniform: EXEC stays all-ones
  int khalf = lane >> 4;
  int row   = tile * 16 + (lane & 15);
  if (row >= nrows) row = nrows - 1;
  const float* xr = X + (size_t)row * HD;

  v8f acc[8];
#pragma unroll
  for (int nt = 0; nt < 8; ++nt)
#pragma unroll
    for (int r = 0; r < 8; ++r) acc[nt][r] = 0.0f;

#pragma unroll
  for (int ks = 0; ks < 4; ++ks) {
    int kb = ks * 32 + khalf * 8;
    float av[16];
#pragma unroll
    for (int j = 0; j < 8; ++j) av[j] = xr[kb + j];
#pragma unroll
    for (int j = 0; j < 8; ++j) av[8 + j] = xr[kb + 16 + j];
    v16bf af;
#pragma unroll
    for (int j = 0; j < 16; ++j) af[j] = (__bf16)av[j];
#pragma unroll
    for (int nt = 0; nt < 8; ++nt) {
      const v16bf bf = *reinterpret_cast<const v16bf*>(
          Bp + (((ks * 8 + nt) * 32 + lane) << 4));
      acc[nt] = __builtin_amdgcn_wmma_f32_16x16x32_bf16(
          false, af, false, bf, (short)0, acc[nt], false, false);
    }
  }

  // C/D layout: VGPR r -> M = r + 8*(lane>>4), N = lane&15 (per 16-col tile)
#pragma unroll
  for (int nt = 0; nt < 8; ++nt)
#pragma unroll
    for (int r = 0; r < 8; ++r) {
      int orow = tile * 16 + r + 8 * khalf;
      if (orow < nrows)
        Hout[(size_t)orow * HD + nt * 16 + (lane & 15)] = acc[nt][r];
    }
}

// alpha_src[i] = h[i].a_src, alpha_dst[i] = h[i].a_dst
__global__ void alpha_kernel(const float* __restrict__ Hm,
                             const float* __restrict__ a_src,
                             const float* __restrict__ a_dst,
                             float* __restrict__ asr, float* __restrict__ ads,
                             int n) {
  int i = blockIdx.x * blockDim.x + threadIdx.x;
  if (i >= n) return;
  const float* hr = Hm + (size_t)i * HD;
  float s = 0.f, d = 0.f;
#pragma unroll 4
  for (int k = 0; k < HD; ++k) {
    float h = hr[k];
    s += h * a_src[k];
    d += h * a_dst[k];
  }
  asr[i] = s;
  ads[i] = d;
}

__global__ void zero_kernel(float* __restrict__ p, int n) {
  int i = blockIdx.x * blockDim.x + threadIdx.x;
  if (i < n) p[i] = 0.0f;
}

// e = leaky_relu(asrc[src]+adst[dst]); segment-max over dst via uint atomicMax
__global__ void edge_logit_kernel(const int* __restrict__ ei, int E, int EE,
                                  const float* __restrict__ asr,
                                  const float* __restrict__ ads,
                                  float* __restrict__ eb,
                                  unsigned* __restrict__ mk) {
  int i = blockIdx.x * blockDim.x + threadIdx.x;
  if (i >= EE) return;
  int s, d;
  if (i < E) { s = ei[i]; d = ei[E + i]; } else { s = i - E; d = s; }
  float e = asr[s] + ads[d];
  e = e > 0.f ? e : NEG_SLOPE * e;
  eb[i] = e;
  atomicMax(mk + d, fenc(e));
}

// ee = exp(e - max[dst]); denom[dst] += ee
__global__ void edge_exp_kernel(const int* __restrict__ ei, int E, int EE,
                                const float* __restrict__ eb,
                                const unsigned* __restrict__ mk,
                                float* __restrict__ eeb,
                                float* __restrict__ den) {
  int i = blockIdx.x * blockDim.x + threadIdx.x;
  if (i >= EE) return;
  int d = (i < E) ? ei[E + i] : (i - E);
  float x = __expf(eb[i] - fdec(mk[d]));
  eeb[i] = x;
  atomicAdd(den + d, x);
}

// out[dst] += h[src] * (ee/denom[dst]); one wave per edge, 4 feats per lane.
// h and out both fit in the 192MB L2 -> atomics are L2-resident.
__global__ void edge_aggr_kernel(const int* __restrict__ ei, int E, int EE,
                                 const float* __restrict__ Hm,
                                 const float* __restrict__ eeb,
                                 const float* __restrict__ den,
                                 float* __restrict__ ag) {
  int t = blockIdx.x * blockDim.x + threadIdx.x;
  int i = t >> 5;
  if (i >= EE) return;
  int fg = (t & 31) << 2;
  int s, d;
  if (i < E) { s = ei[i]; d = ei[E + i]; } else { s = i - E; d = s; }
  float alpha = eeb[i] / den[d];
  const float4 hv =
      *reinterpret_cast<const float4*>(Hm + (size_t)s * HD + fg);
  float* o = ag + (size_t)d * HD + fg;
  atomicAdd(o + 0, hv.x * alpha);
  atomicAdd(o + 1, hv.y * alpha);
  atomicAdd(o + 2, hv.z * alpha);
  atomicAdd(o + 3, hv.w * alpha);
}

__global__ void bias_relu_kernel(const float* __restrict__ ag,
                                 const float* __restrict__ b,
                                 float* __restrict__ xo, int nh) {
  int i = blockIdx.x * blockDim.x + threadIdx.x;
  if (i >= nh) return;
  float v = ag[i] + b[i & (HD - 1)];
  xo[i] = v > 0.f ? v : 0.f;
}

__global__ void pool_acc_kernel(const float* __restrict__ x,
                                const int* __restrict__ batch,
                                float* __restrict__ pooled,
                                float* __restrict__ counts, int n) {
  int t = blockIdx.x * blockDim.x + threadIdx.x;
  int i = t >> 5;
  if (i >= n) return;
  int fg = (t & 31) << 2;
  int g = batch[i];
  const float4 v = *reinterpret_cast<const float4*>(x + (size_t)i * HD + fg);
  float* o = pooled + g * HD + fg;
  atomicAdd(o + 0, v.x);
  atomicAdd(o + 1, v.y);
  atomicAdd(o + 2, v.z);
  atomicAdd(o + 3, v.w);
  if (fg == 0) atomicAdd(counts + g, 1.0f);
}

__global__ void pool_div_kernel(float* __restrict__ pooled,
                                const float* __restrict__ counts) {
  int t = blockIdx.x * blockDim.x + threadIdx.x;
  if (t >= GNUM * HD) return;
  pooled[t] /= fmaxf(counts[t >> 7], 1.0f);
}

__global__ void doc_kernel(const float* __restrict__ doc,
                           const float* __restrict__ Wd,
                           const float* __restrict__ bd,
                           float* __restrict__ de) {
  int t = blockIdx.x * blockDim.x + threadIdx.x;
  if (t >= GNUM * HD) return;
  int g = t >> 7, h = t & (HD - 1);
  float s = bd[h];
  const float* dr = doc + (size_t)g * DDOCD;
#pragma unroll 4
  for (int k = 0; k < DDOCD; ++k) s += dr[k] * Wd[k * HD + h];
  de[t] = s > 0.f ? s : 0.f;
}

__global__ void heads_kernel(const float* __restrict__ pooled,
                             const float* __restrict__ de,
                             const float* __restrict__ Wt,
                             const float* __restrict__ bt,
                             const float* __restrict__ Wm,
                             const float* __restrict__ bm,
                             float* __restrict__ out) {
  int t = blockIdx.x * blockDim.x + threadIdx.x;
  if (t >= GNUM * (TOUTD + 1)) return;
  int g = t / (TOUTD + 1), o = t % (TOUTD + 1);
  if (o < TOUTD) {
    float s = bt[o];
    for (int k = 0; k < HD; ++k) s += pooled[g * HD + k] * Wt[k * TOUTD + o];
    for (int k = 0; k < HD; ++k) s += de[g * HD + k] * Wt[(HD + k) * TOUTD + o];
    out[g * TOUTD + o] = s;
  } else {
    float s = bm[0];
    for (int k = 0; k < HD; ++k) s += pooled[g * HD + k] * Wm[k];
    for (int k = 0; k < HD; ++k) s += de[g * HD + k] * Wm[HD + k];
    out[GNUM * TOUTD + g] = s;
  }
}

extern "C" void kernel_launch(void* const* d_in, const int* in_sizes, int n_in,
                              void* d_out, int out_size, void* d_ws,
                              size_t ws_size, hipStream_t stream) {
  (void)n_in; (void)out_size; (void)ws_size;
  const float* x0    = (const float*)d_in[0];
  const int*   ei    = (const int*)d_in[1];
  const int*   batch = (const int*)d_in[2];
  const float* doc   = (const float*)d_in[3];
  const float* Wl[3]  = {(const float*)d_in[4],  (const float*)d_in[8],  (const float*)d_in[12]};
  const float* asl[3] = {(const float*)d_in[5],  (const float*)d_in[9],  (const float*)d_in[13]};
  const float* adl[3] = {(const float*)d_in[6],  (const float*)d_in[10], (const float*)d_in[14]};
  const float* bl[3]  = {(const float*)d_in[7],  (const float*)d_in[11], (const float*)d_in[15]};
  const float* Wdoc = (const float*)d_in[16];
  const float* bdoc = (const float*)d_in[17];
  const float* Wt   = (const float*)d_in[18];
  const float* bt   = (const float*)d_in[19];
  const float* Wm   = (const float*)d_in[20];
  const float* bm   = (const float*)d_in[21];
  float* out = (float*)d_out;

  const int N  = in_sizes[0] / HD;
  const int E  = in_sizes[1] / 2;
  const int EE = E + N;
  const size_t NH = (size_t)N * HD;

  float*    ws  = (float*)d_ws;
  float*    xA  = ws;                 // layer activations  [N*128]
  float*    hb  = ws + NH;            // h = x@W            [N*128]
  float*    ag  = ws + 2 * NH;        // aggregation accum  [N*128]
  float*    asr = ws + 3 * NH;        // alpha_src          [N]
  float*    ads = asr + N;            // alpha_dst          [N]
  unsigned* mk  = (unsigned*)(ads + N);  // seg-max keys    [N]
  float*    den = ads + 2 * N;        // softmax denom      [N]
  float*    eb  = den + N;            // edge logits        [EE]
  float*    eeb = eb + EE;            // edge exp           [EE]
  __bf16*   wp  = (__bf16*)(eeb + EE);   // packed weights  [3*16384]
  float*    pooled = (float*)(wp + 3 * 16384);  // [G*128]
  float*    counts = pooled + GNUM * HD;        // [G]
  float*    de     = counts + GNUM;             // [G*128]

  const int T = 256;

  for (int l = 0; l < 3; ++l)
    pack_w_kernel<<<(16384 + T - 1) / T, T, 0, stream>>>(Wl[l],
                                                         wp + (size_t)l * 16384);

  const int ntiles  = (N + 15) / 16;
  const int gblocks = (ntiles + 7) / 8;

  const float* xin = x0;
  for (int l = 0; l < 3; ++l) {
    gemm128_wmma_kernel<<<gblocks, 256, 0, stream>>>(xin, wp + (size_t)l * 16384,
                                                     hb, N);
    alpha_kernel<<<(N + T - 1) / T, T, 0, stream>>>(hb, asl[l], adl[l], asr,
                                                    ads, N);
    zero_kernel<<<((int)NH + T - 1) / T, T, 0, stream>>>(ag, (int)NH);
    zero_kernel<<<(2 * N + T - 1) / T, T, 0, stream>>>(ads + N, 2 * N);  // mk+den
    edge_logit_kernel<<<(EE + T - 1) / T, T, 0, stream>>>(ei, E, EE, asr, ads,
                                                          eb, mk);
    edge_exp_kernel<<<(EE + T - 1) / T, T, 0, stream>>>(ei, E, EE, eb, mk, eeb,
                                                        den);
    const int aggT = EE * 32;
    edge_aggr_kernel<<<(aggT + T - 1) / T, T, 0, stream>>>(ei, E, EE, hb, eeb,
                                                           den, ag);
    bias_relu_kernel<<<((int)NH + T - 1) / T, T, 0, stream>>>(ag, bl[l], xA,
                                                              (int)NH);
    xin = xA;
  }

  zero_kernel<<<(GNUM * HD + GNUM + T - 1) / T, T, 0, stream>>>(
      pooled, GNUM * HD + GNUM);
  pool_acc_kernel<<<(N * 32 + T - 1) / T, T, 0, stream>>>(xA, batch, pooled,
                                                          counts, N);
  pool_div_kernel<<<(GNUM * HD + T - 1) / T, T, 0, stream>>>(pooled, counts);
  doc_kernel<<<(GNUM * HD + T - 1) / T, T, 0, stream>>>(doc, Wdoc, bdoc, de);
  heads_kernel<<<(GNUM * (TOUTD + 1) + T - 1) / T, T, 0, stream>>>(
      pooled, de, Wt, bt, Wm, bm, out);
}
